// EncoderV2_21174188769500
// MI455X (gfx1250) — matrix operand order repaired and checked by reading, not compiled, not run on hardware
//
#include <hip/hip_runtime.h>
#include <hip/hip_bf16.h>
#include <stdint.h>

#define N_NODES   100000
#define N_EDGES   1600000
#define N_GRAPHS  512
#define D         50
#define DE        8
#define DZ        108      // 2*D + DE
#define KP        128      // K padded for WMMA (4 x 32)
#define N_OUT     10
#define MT        2        // M-tiles (of 16 edges) per wave
#define EPW       (16*MT)  // 32 edges per wave
#define EPB       (8*EPW)  // 256 edges per block (8 waves)

typedef __attribute__((ext_vector_type(16))) __bf16 v16bf;
typedef __attribute__((ext_vector_type(8)))  float  v8f;

union Frag16 {
    v16bf    v;
    uint4    q[2];
};

__device__ __forceinline__ unsigned short f2bf(float f) {
    uint32_t u = __float_as_uint(f);
    u += 0x7FFFu + ((u >> 16) & 1u);   // round-to-nearest-even
    return (unsigned short)(u >> 16);
}

// packed f32x2 -> bf16x2 (lowers to v_cvt_pk_bf16_f32)
__device__ __forceinline__ uint32_t pack2bf(float a, float b) {
    union { __hip_bfloat162 h; uint32_t u; } cv;
    cv.h = __float22bfloat162_rn(float2{a, b});
    return cv.u;
}

// ---------------------------------------------------------------------------
// Edge kernel: gather z = [h[dst], h[src], e], two bf16 WMMA GEMMs against
// Wf/Ws staged in LDS, branchless sigmoid*softplus gate, atomic scatter-add.
// Fragments are loaded as explicit uint4 pairs (forces ds_load_b128), and all
// global offsets are 32-bit so atomics/loads use saddr + scaled voffset.
// ---------------------------------------------------------------------------
__global__ __launch_bounds__(256) void edge_wmma_kernel(
    const float* __restrict__ h,
    const int*   __restrict__ edge_index,   // [2, E]: row0=src, row1=dst
    const float* __restrict__ ea,           // [E, DE]
    const float* __restrict__ Wf, const float* __restrict__ bfv,
    const float* __restrict__ Ws, const float* __restrict__ bsv,
    float*       __restrict__ hout,         // pre-initialized with residual h
    int layer, int nblocks)
{
    __shared__ unsigned short sWf[64 * KP];       // [n][k], bf16, zero padded
    __shared__ unsigned short sWs[64 * KP];
    __shared__ unsigned short sZ[8][EPW * KP];    // per-wave z tiles [m][k]
    __shared__ int sSrc[EPB];
    __shared__ int sDst[EPB];

    const int tid  = threadIdx.x;
    const int wave = tid >> 5;
    const int lane = tid & 31;
    const int e0   = (int)blockIdx.x * EPB;       // < 1.6M, fits int

    // ---- prefetch next block's edge data into L2 (global_prefetch_b8) ----
    if (blockIdx.x + 1 < (unsigned)nblocks) {
        int en = e0 + EPB;
        __builtin_prefetch(edge_index + en + tid, 0, 1);                 // src ids
        __builtin_prefetch(edge_index + N_EDGES + en + tid, 0, 1);       // dst ids
        __builtin_prefetch(ea + (size_t)en * DE + tid * 8, 0, 1);        // attrs
    }

    // ---- stage weights for this layer into LDS (shared by all 8 waves) ----
    const float* Wf0 = Wf + layer * (D * DZ);
    const float* Ws0 = Ws + layer * (D * DZ);
    for (int idx = tid; idx < 64 * KP; idx += 256) {
        int n = idx >> 7, k = idx & (KP - 1);
        bool ok = (n < D) && (k < DZ);
        sWf[idx] = f2bf(ok ? Wf0[n * DZ + k] : 0.0f);
        sWs[idx] = f2bf(ok ? Ws0[n * DZ + k] : 0.0f);
    }
    // ---- stage edge endpoints ----
    if (tid < EPB) {
        int e = e0 + tid;
        sSrc[tid] = edge_index[e];
        sDst[tid] = edge_index[N_EDGES + e];
    }
    __syncthreads();

    // ---- build this wave's 32x128 bf16 z tile (pair-packed stores) ----
    for (int r = 0; r < EPW; ++r) {
        int er = wave * EPW + r;
        int e  = e0 + er;
        int od = sDst[er] * D;        // 32-bit offsets (< 5M)
        int os = sSrc[er] * D;
        int oe = e * DE;
        uint32_t* zo = (uint32_t*)(&sZ[wave][r * KP]);
        for (int kk = lane * 2; kk < KP; kk += 64) {
            float v0, v1;
            if      (kk < D)      { v0 = h[od + kk];           v1 = h[od + kk + 1]; }
            else if (kk < 2 * D)  { v0 = h[os + kk - D];       v1 = h[os + kk - D + 1]; }
            else if (kk < DZ)     { v0 = ea[oe + kk - 2 * D];  v1 = ea[oe + kk - 2 * D + 1]; }
            else                  { v0 = 0.0f;                 v1 = 0.0f; }
            zo[kk >> 1] = pack2bf(v0, v1);
        }
    }
    __syncthreads();

    const int m  = lane & 15;   // A-row / B-column / D-column within tile
    const int hi = lane >> 4;

    // ---- per M-tile: hoisted A frags, then 4 N-tiles x 4 K-steps x 2 ----
#pragma unroll
    for (int mt = 0; mt < MT; ++mt) {
        // A fragments (ISA 16-bit A layout) as explicit b128 pairs:
        // per K-block b: quads at uint index b*16 + {hi*4, 8 + hi*4}
        const uint4* zq = (const uint4*)(&sZ[wave][(mt * 16 + m) * KP]);
        Frag16 a[4];
#pragma unroll
        for (int b = 0; b < 4; ++b) {
            a[b].q[0] = zq[b * 4 + hi];
            a[b].q[1] = zq[b * 4 + 2 + hi];
        }

#pragma unroll
        for (int nt = 0; nt < 4; ++nt) {
            v8f accf = {};
            v8f accs = {};
            // B fragments: quads at uint index b*16 + {hi*4, 8 + hi*4}
            const uint4* wfq = (const uint4*)(&sWf[(nt * 16 + m) * KP]);
            const uint4* wsq = (const uint4*)(&sWs[(nt * 16 + m) * KP]);
#pragma unroll
            for (int b = 0; b < 4; ++b) {
                Frag16 bfF, bsF;
                bfF.q[0] = wfq[b * 4 + hi];
                bfF.q[1] = wfq[b * 4 + 2 + hi];
                bsF.q[0] = wsq[b * 4 + hi];
                bsF.q[1] = wsq[b * 4 + 2 + hi];
                accf = __builtin_amdgcn_wmma_f32_16x16x32_bf16(
                    false, a[b].v, false, bfF.v, (short)0, accf, false, false);
                accs = __builtin_amdgcn_wmma_f32_16x16x32_bf16(
                    false, a[b].v, false, bsF.v, (short)0, accs, false, false);
            }
            int n = nt * 16 + m;  // output channel
            if (n < D) {
                float bF = bfv[layer * D + n];
                float bS = bsv[layer * D + n];
#pragma unroll
                for (int v = 0; v < 8; ++v) {
                    int row  = mt * 16 + hi * 8 + v;   // edge row within wave
                    float zf = accf[v] + bF;
                    float zs = accs[v] + bS;
                    // branchless sigmoid and softplus (v_exp/v_log only)
                    float gate = 1.0f / (1.0f + __expf(-zf));
                    float sp   = fmaxf(zs, 0.0f)
                               + __logf(1.0f + __expf(-fabsf(zs)));
                    int off = sDst[wave * EPW + row] * D + n;  // 32-bit offset
                    atomicAdd(&hout[off], gate * sp);
                }
            }
        }
    }
}

// ---------------------------------------------------------------------------
// Helpers
// ---------------------------------------------------------------------------
__global__ void copy_kernel(const float* __restrict__ src,
                            float* __restrict__ dst, long n) {
    long i = (long)blockIdx.x * blockDim.x + threadIdx.x;
    if (i < n) dst[i] = src[i];
}

__global__ void zero_kernel(float* __restrict__ p, long n) {
    long i = (long)blockIdx.x * blockDim.x + threadIdx.x;
    if (i < n) p[i] = 0.0f;
}

__global__ void count_kernel(const int* __restrict__ batch,
                             float* __restrict__ cnt) {
    int v = blockIdx.x * blockDim.x + threadIdx.x;
    if (v < N_NODES) atomicAdd(&cnt[batch[v]], 1.0f);
}

// ReLU in place + segment max / segment sum into pool buffers (values >= 0
// after relu, so int-bit atomicMax == float max; mx is zero-initialized).
__global__ void relu_pool_kernel(float* __restrict__ h,
                                 const int* __restrict__ batch,
                                 float* __restrict__ mx,
                                 float* __restrict__ sm) {
    long i = (long)blockIdx.x * blockDim.x + threadIdx.x;
    if (i >= (long)N_NODES * D) return;
    int node = (int)(i / D), c = (int)(i % D);
    float v = h[i];
    v = v > 0.0f ? v : 0.0f;
    h[i] = v;
    int g = batch[node];
    atomicMax((int*)&mx[g * D + c], __float_as_int(v));
    atomicAdd(&sm[g * D + c], v);
}

__global__ void pool_acc_kernel(const float* __restrict__ mx,
                                const float* __restrict__ sm,
                                const float* __restrict__ cnt,
                                float* __restrict__ enc) {
    int i = blockIdx.x * blockDim.x + threadIdx.x;
    if (i >= N_GRAPHS * D) return;
    int g = i / D, c = i % D;
    float cn = cnt[g];
    cn = cn > 1.0f ? cn : 1.0f;
    enc[g * 2 * D + c]     += mx[i];
    enc[g * 2 * D + D + c] += sm[i] / cn;
}

__global__ void head_kernel(const float* __restrict__ enc,
                            const float* __restrict__ lw,
                            const float* __restrict__ lb,
                            float* __restrict__ out) {
    int g = blockIdx.x * blockDim.x + threadIdx.x;
    if (g >= N_GRAPHS) return;
    const float* er = enc + g * 2 * D;
    float l[N_OUT];
    float mxv = -1e30f;
#pragma unroll
    for (int j = 0; j < N_OUT; ++j) {
        float s = lb[j];
        for (int c = 0; c < 2 * D; ++c) s += er[c] * lw[j * 2 * D + c];
        l[j] = s;
        mxv = fmaxf(mxv, s);
    }
    float se = 0.0f;
#pragma unroll
    for (int j = 0; j < N_OUT; ++j) se += __expf(l[j] - mxv);
    float lse = mxv + __logf(se);
#pragma unroll
    for (int j = 0; j < N_OUT; ++j) out[g * N_OUT + j] = l[j] - lse;
    float* o2 = out + N_GRAPHS * N_OUT;
    for (int c = 0; c < 2 * D; ++c) o2[g * 2 * D + c] = er[c];
}

// ---------------------------------------------------------------------------
// Orchestration
// ---------------------------------------------------------------------------
extern "C" void kernel_launch(void* const* d_in, const int* in_sizes, int n_in,
                              void* d_out, int out_size, void* d_ws, size_t ws_size,
                              hipStream_t stream) {
    const float* x    = (const float*)d_in[0];
    const int*   ei   = (const int*)  d_in[1];
    const float* eatt = (const float*)d_in[2];
    const int*   bat  = (const int*)  d_in[3];
    const float* Wf   = (const float*)d_in[4];
    const float* bf   = (const float*)d_in[5];
    const float* Ws   = (const float*)d_in[6];
    const float* bs   = (const float*)d_in[7];
    const float* lw   = (const float*)d_in[8];
    const float* lb   = (const float*)d_in[9];
    float* out = (float*)d_out;

    const long HN = (long)N_NODES * D;           // 5,000,000
    float* ws  = (float*)d_ws;
    float* h0  = ws;                             // HN
    float* h1  = h0 + HN;                        // HN
    float* cnt = h1 + HN;                        // N_GRAPHS
    float* mx  = cnt + N_GRAPHS;                 // N_GRAPHS * D
    float* sm  = mx + (long)N_GRAPHS * D;        // N_GRAPHS * D
    float* enc = sm + (long)N_GRAPHS * D;        // N_GRAPHS * 2D

    const int T = 256;
    const int gH = (int)((HN + T - 1) / T);

    // h0 = x ; zero cnt..enc block ; compute per-graph node counts
    copy_kernel<<<gH, T, 0, stream>>>(x, h0, HN);
    long zn = (long)N_GRAPHS * (1 + D + D + 2 * D);
    zero_kernel<<<(int)((zn + T - 1) / T), T, 0, stream>>>(cnt, zn);
    count_kernel<<<(N_NODES + T - 1) / T, T, 0, stream>>>(bat, cnt);

    float* cur = h0;
    float* nxt = h1;
    const int edgeBlocks = N_EDGES / EPB;        // 6250
    for (int layer = 0; layer < 3; ++layer) {
        long pn = (long)N_GRAPHS * D * 2;        // mx and sm are contiguous
        zero_kernel<<<(int)((pn + T - 1) / T), T, 0, stream>>>(mx, pn);
        copy_kernel<<<gH, T, 0, stream>>>(cur, nxt, HN);  // residual init
        edge_wmma_kernel<<<edgeBlocks, 256, 0, stream>>>(
            cur, ei, eatt, Wf, bf, Ws, bs, nxt, layer, edgeBlocks);
        relu_pool_kernel<<<gH, T, 0, stream>>>(nxt, bat, mx, sm);
        pool_acc_kernel<<<(N_GRAPHS * D + T - 1) / T, T, 0, stream>>>(mx, sm, cnt, enc);
        float* t = cur; cur = nxt; nxt = t;
    }

    head_kernel<<<(N_GRAPHS + T - 1) / T, T, 0, stream>>>(enc, lw, lb, out);
}
